// DeformableConvLayer_53077205844107
// MI455X (gfx1250) — matrix operand order repaired and checked by reading, not compiled, not run on hardware
//
#include <hip/hip_runtime.h>
#include <cstddef>

// ---------------------------------------------------------------------------
// DeformableConvLayer for MI455X (gfx1250, wave32)
//
// Pipeline:
//   k_fold  : xf[b,h,w] = sum_c x*w_icfd + b_icfd   (128 MiB read, BW-bound)
//             + deterministic per-block partial sums for the mean
//   k_taps  : per batch: mean -> 1600 (dy,dx) -> 6400 bilinear taps binned
//             into dense 33x33 kernel (radius 16) + overflow list; compacted
//             deterministically.
//   k_accum : inp = xf + 64*b_fus + sparse-kernel correlation of xf,
//             via 32x32 LDS tile with 16-px halo (64x64 floats = 16 KB LDS).
//   k_conv  : 3x3 conv 1->64 channels as WMMA GEMM. One wave per 16-pixel
//             tile: B (patch, f16) built once with clamp+select (no exec
//             branching), then 4x v_wmma_f32_16x16x32_f16 over the 4 oc-tiles
//             with bias preloaded into the C accumulator.
// ---------------------------------------------------------------------------

#define BB 8
#define CC 64
#define HH 256
#define WW 256
#define HWP (HH*WW)          // 65536
#define GG 25
#define DFC 64
#define OCN 64
#define RAD 16
#define BINW 33              // 2*RAD+1
#define NBINS (BINW*BINW)    // 1089
#define TCAP 1200
#define OCAP 6400

typedef __attribute__((ext_vector_type(16))) _Float16 v16h;
typedef __attribute__((ext_vector_type(8)))  float    v8f;

// ---------------- Kernel 1: channel fold + partial sums --------------------
__global__ __launch_bounds__(256)
void k_fold(const float* __restrict__ x, const float* __restrict__ w_icfd,
            const float* __restrict__ b_icfd,
            float* __restrict__ xf, float* __restrict__ partials) {
    const int b   = blockIdx.y;
    const int g4  = blockIdx.x * 256 + threadIdx.x;   // float4 group id
    const int q   = HWP / 4;                          // 16384 groups / image
    const float4* xv = (const float4*)x;
    size_t base = (size_t)b * CC * q + g4;

    float4 acc; acc.x = acc.y = acc.z = acc.w = 0.0f;
#pragma unroll 4
    for (int c = 0; c < CC; ++c) {
        float4 v = xv[base + (size_t)c * q];
        float wc = w_icfd[c];                          // uniform -> s_load
        acc.x += v.x * wc; acc.y += v.y * wc;
        acc.z += v.z * wc; acc.w += v.w * wc;
    }
    float bi = b_icfd[0];
    acc.x += bi; acc.y += bi; acc.z += bi; acc.w += bi;
    ((float4*)xf)[(size_t)b * q + g4] = acc;

    // deterministic block reduction for mean
    __shared__ float red[256];
    red[threadIdx.x] = acc.x + acc.y + acc.z + acc.w;
    __syncthreads();
#pragma unroll
    for (int s = 128; s > 0; s >>= 1) {
        if (threadIdx.x < s) red[threadIdx.x] += red[threadIdx.x + s];
        __syncthreads();
    }
    if (threadIdx.x == 0) partials[b * 64 + blockIdx.x] = red[0];
}

// ---------------- Kernel 2: offsets -> binned tap kernel -------------------
__global__ __launch_bounds__(256)
void k_taps(const float* __restrict__ w_off, const float* __restrict__ b_off,
            const float* __restrict__ w_fus,
            const float* __restrict__ partials,
            int* __restrict__ tY, int* __restrict__ tX, float* __restrict__ tW,
            int* __restrict__ oY, int* __restrict__ oX, float* __restrict__ oW,
            int* __restrict__ counts) {
    const int b   = blockIdx.x;
    const int tid = threadIdx.x;

    __shared__ float red[64];
    __shared__ float bins[NBINS];
    __shared__ float meanS;

    if (tid < 64) red[tid] = partials[b * 64 + tid];
    __syncthreads();
#pragma unroll
    for (int s = 32; s > 0; s >>= 1) {
        if (tid < s) red[tid] += red[tid + s];
        __syncthreads();
    }
    if (tid == 0) meanS = red[0] / (float)HWP;
    for (int i = tid; i < NBINS; i += 256) bins[i] = 0.0f;
    __syncthreads();

    const float mean = meanS;
    int novl = 0;  // thread 0 only

    // every thread walks all 1600 entries (uniform scalar loads of w_off/b_off);
    // each thread accumulates only the bins it owns -> deterministic order.
    for (int e = 0; e < GG * DFC; ++e) {
        const int g = e / DFC;
        const int k = e * 2;
        float dy = mean * w_off[k]     + b_off[k];
        float dx = mean * w_off[k + 1] + b_off[k + 1];
        float syf = floorf(dy), sxf = floorf(dx);
        float fy = dy - syf, fx = dx - sxf;
        int   sy = (int)syf, sx = (int)sxf;
        float wg = w_fus[g];

        int   ty4[4] = { sy, sy,     sy + 1, sy + 1 };
        int   tx4[4] = { sx, sx + 1, sx,     sx + 1 };
        float tw4[4] = { wg * (1.f - fy) * (1.f - fx), wg * (1.f - fy) * fx,
                         wg * fy * (1.f - fx),         wg * fy * fx };
#pragma unroll
        for (int t = 0; t < 4; ++t) {
            int ty_ = ty4[t], tx_ = tx4[t];
            if (ty_ >= -RAD && ty_ <= RAD && tx_ >= -RAD && tx_ <= RAD) {
                int bin = (ty_ + RAD) * BINW + (tx_ + RAD);
                if ((bin & 255) == tid) bins[bin] += tw4[t];
            } else if (tid == 0) {           // serial, deterministic overflow
                oY[b * OCAP + novl] = ty_;
                oX[b * OCAP + novl] = tx_;
                oW[b * OCAP + novl] = tw4[t];
                ++novl;
            }
        }
    }
    __syncthreads();

    if (tid == 0) {                          // deterministic compaction
        int nt = 0;
        for (int i = 0; i < NBINS; ++i) {
            float w = bins[i];
            if (w != 0.0f) {
                tY[b * TCAP + nt] = i / BINW - RAD;
                tX[b * TCAP + nt] = i % BINW - RAD;
                tW[b * TCAP + nt] = w;
                ++nt;
            }
        }
        counts[b * 2]     = nt;
        counts[b * 2 + 1] = novl;
    }
}

// ---------------- Kernel 3: sparse-kernel correlation via LDS tile ---------
__global__ __launch_bounds__(256)
void k_accum(const float* __restrict__ xf, const float* __restrict__ b_fus,
             const int* __restrict__ tY, const int* __restrict__ tX,
             const float* __restrict__ tW,
             const int* __restrict__ oY, const int* __restrict__ oX,
             const float* __restrict__ oW,
             const int* __restrict__ counts, float* __restrict__ inp) {
    const int b  = blockIdx.z;
    const int w0 = blockIdx.x * 32;
    const int h0 = blockIdx.y * 32;
    const int lx = threadIdx.x;              // 0..31
    const int ly = threadIdx.y;              // 0..7
    const int tid = ly * 32 + lx;

    __shared__ float sm[64 * 64];            // 32x32 tile + 16-px halo
    const float* xfb = xf + (size_t)b * HWP;

    for (int l = tid; l < 64 * 64; l += 256) {
        int gy = (l >> 6) - RAD + h0;
        int gx = (l & 63) - RAD + w0;
        sm[l] = (gy >= 0 && gy < HH && gx >= 0 && gx < WW)
                    ? xfb[gy * WW + gx] : 0.0f;
    }
    __syncthreads();

    const int nt  = counts[b * 2];
    const int nov = counts[b * 2 + 1];

    float a0 = 0.f, a1 = 0.f, a2 = 0.f, a3 = 0.f;
    for (int t = 0; t < nt; ++t) {
        int   sy = tY[b * TCAP + t];         // uniform -> scalar loads
        int   sx = tX[b * TCAP + t];
        float tw = tW[b * TCAP + t];
        int idx = (ly + RAD + sy) * 64 + (lx + RAD + sx);
        a0 += tw * sm[idx];
        a1 += tw * sm[idx + 8 * 64];
        a2 += tw * sm[idx + 16 * 64];
        a3 += tw * sm[idx + 24 * 64];
    }
    for (int t = 0; t < nov; ++t) {          // rare fallback, global gather
        int   sy = oY[b * OCAP + t];
        int   sx = oX[b * OCAP + t];
        float tw = oW[b * OCAP + t];
        int gx = w0 + lx + sx;
        bool vx = (gx >= 0 && gx < WW);
#pragma unroll
        for (int j = 0; j < 4; ++j) {
            int gy = h0 + ly + 8 * j + sy;
            float v = (vx && gy >= 0 && gy < HH) ? xfb[gy * WW + gx] : 0.0f;
            float* aj = (j == 0) ? &a0 : (j == 1) ? &a1 : (j == 2) ? &a2 : &a3;
            *aj += tw * v;
        }
    }

    const float add = (float)DFC * b_fus[0];
    float acc[4] = { a0, a1, a2, a3 };
#pragma unroll
    for (int j = 0; j < 4; ++j) {
        int hy = ly + 8 * j;
        float center = sm[(hy + RAD) * 64 + (lx + RAD)];
        inp[(size_t)b * HWP + (size_t)(h0 + hy) * WW + (w0 + lx)] =
            acc[j] + center + add;
    }
}

// ---------------- Kernel 4: 3x3 conv 1->64 via WMMA f16 --------------------
// One wave per 16-pixel tile; B built once (clamp + select, no branches);
// 4 WMMAs over the 4 oc-tiles with bias preloaded into C.
__global__ __launch_bounds__(256)
void k_conv(const float* __restrict__ inp, const float* __restrict__ w_conv,
            const float* __restrict__ b_conv, float* __restrict__ out) {
    const int tid   = threadIdx.x;
    const int wv    = tid >> 5;              // 8 waves / block
    const int lane  = tid & 31;
    const int half  = lane >> 4;             // 0: lanes 0-15, 1: lanes 16-31
    const int lhalf = lane & 15;
    const int pxt   = blockIdx.x * 8 + wv;   // 0..32767 pixel tiles

    const int b  = pxt >> 12;                // 4096 tiles per image
    const int pt = pxt & 4095;
    const int h  = pt >> 4;
    const int w0 = (pt & 15) << 4;

    // B: 32x16 f16, columns = 16 pixels along w, rows = 3x3 patch taps.
    // Lanes 0-15 hold K=0..15 of column N=lane; lanes 16-31 hold K=16..31 (0).
    // Clamp the address (always legal) and select 0 for out-of-bounds /
    // upper-half lanes -> plain loads + v_cndmask, no exec manipulation.
    v16h bv = (v16h)(_Float16)0.0f;
    {
        const float* ib = inp + (size_t)b * HWP;
        const int n = lhalf;
#pragma unroll
        for (int k = 0; k < 9; ++k) {
            const int ky = k / 3 - 1, kx = k % 3 - 1;
            int yy = h + ky, xx = w0 + n + kx;
            bool ok = (yy >= 0) & (yy < HH) & (xx >= 0) & (xx < WW) & (half == 0);
            int yc = yy < 0 ? 0 : (yy > HH - 1 ? HH - 1 : yy);
            int xc = xx < 0 ? 0 : (xx > WW - 1 ? WW - 1 : xx);
            float v = ib[yc * WW + xc];
            bv[k] = ok ? (_Float16)v : (_Float16)0.0f;
        }
    }

    const int N = lhalf;
    const int Mbase = half ? 8 : 0;

#pragma unroll
    for (int octile = 0; octile < 4; ++octile) {
        // A: 16x32 f16, rows = output channels, K = 9 real taps (rest 0).
        // Lanes 0-15 hold K=0..15 of row M=lane; lanes 16-31 hold
        // K=16..31 (0) of row M=lane-16 -> K=8 lives in lower lanes, e8.
        const int oc_a = octile * 16 + lhalf;
        float wk[9];
#pragma unroll
        for (int k = 0; k < 9; ++k) wk[k] = w_conv[oc_a * 9 + k];

        v16h av = (v16h)(_Float16)0.0f;
        av[0] = half ? (_Float16)0.0f : (_Float16)wk[0];
#pragma unroll
        for (int k = 1; k < 9; ++k)
            av[k] = half ? (_Float16)0.0f : (_Float16)wk[k];

        // C preloaded with bias: D = A*B + bias in one WMMA.
        v8f c;
#pragma unroll
        for (int i = 0; i < 8; ++i) c[i] = b_conv[octile * 16 + Mbase + i];

        c = __builtin_amdgcn_wmma_f32_16x16x32_f16(
                /*neg_a=*/false, av, /*neg_b=*/false, bv,
                /*c_mod=*/(short)0, c, /*reuse_a=*/false, /*reuse_b=*/false);

        // D layout: lanes 0-15 -> N=lane, rows M=0..7 in c[0..7];
        //           lanes 16-31 -> N=lane-16, rows M=8..15 in c[0..7].
#pragma unroll
        for (int i = 0; i < 8; ++i) {
            int oc = octile * 16 + Mbase + i;
            out[((size_t)(b * OCN + oc)) * HWP + (size_t)h * WW + w0 + N] = c[i];
        }
    }
}

// ---------------------------- launcher -------------------------------------
extern "C" void kernel_launch(void* const* d_in, const int* in_sizes, int n_in,
                              void* d_out, int out_size, void* d_ws, size_t ws_size,
                              hipStream_t stream) {
    const float* x      = (const float*)d_in[0];
    const float* w_icfd = (const float*)d_in[1];
    const float* b_icfd = (const float*)d_in[2];
    const float* w_off  = (const float*)d_in[3];
    const float* b_off  = (const float*)d_in[4];
    const float* w_fus  = (const float*)d_in[5];
    const float* b_fus  = (const float*)d_in[6];
    const float* w_conv = (const float*)d_in[7];
    const float* b_conv = (const float*)d_in[8];
    float* out = (float*)d_out;

    float* ws       = (float*)d_ws;
    float* xf       = ws;                         // B*HW
    float* inp      = xf  + (size_t)BB * HWP;     // B*HW
    float* partials = inp + (size_t)BB * HWP;     // B*64
    int*   counts   = (int*)(partials + BB * 64); // B*2 (pad 16)
    int*   tY       = counts + 16;
    int*   tX       = tY + BB * TCAP;
    float* tW       = (float*)(tX + BB * TCAP);
    int*   oY       = (int*)(tW + BB * TCAP);
    int*   oX       = oY + BB * OCAP;
    float* oW       = (float*)(oX + BB * OCAP);

    k_fold<<<dim3(64, BB), 256, 0, stream>>>(x, w_icfd, b_icfd, xf, partials);
    k_taps<<<BB, 256, 0, stream>>>(w_off, b_off, w_fus, partials,
                                   tY, tX, tW, oY, oX, oW, counts);
    k_accum<<<dim3(8, 8, BB), dim3(32, 8), 0, stream>>>(
        xf, b_fus, tY, tX, tW, oY, oX, oW, counts, inp);
    k_conv<<<(BB * (HWP / 16)) / 8, 256, 0, stream>>>(inp, w_conv, b_conv, out);
}
